// PerNodeViewEncoder_76836964926321
// MI455X (gfx1250) — matrix operand order repaired and checked by reading, not compiled, not run on hardware
//
#include <hip/hip_runtime.h>
#include <hip/hip_bf16.h>

// ---------------------------------------------------------------------------
// PerNodeViewEncoder for MI455X (gfx1250, wave32, WMMA bf16 w/ f32 accumulate)
// ---------------------------------------------------------------------------
#define NN      50000
#define DD      128
#define EDIM    32
#define TDIM    128
#define ETOT    1000000
#define EV      250000
#define KVIEWS  4
#define LAYERS  2

typedef __attribute__((ext_vector_type(16))) __bf16         v16bf;
typedef __attribute__((ext_vector_type(8)))  float          v8f;
typedef __attribute__((ext_vector_type(8)))  unsigned short u16x8;
typedef __attribute__((ext_vector_type(16))) unsigned short u16x16;

__device__ __forceinline__ unsigned short f2bf(float f) {
    unsigned int u = __builtin_bit_cast(unsigned int, f);
    unsigned int r = u + 0x7FFFu + ((u >> 16) & 1u);   // round-to-nearest-even
    return (unsigned short)(r >> 16);
}

__device__ __forceinline__ v16bf frag_from_chunks(u16x8 lo, u16x8 hi) {
    union { u16x16 u; v16bf b; } c;
#pragma unroll
    for (int i = 0; i < 8; ++i) { c.u[i] = lo[i]; c.u[i + 8] = hi[i]; }
    return c.b;
}
__device__ __forceinline__ v16bf frag_from16(const unsigned short* p) {
    union { u16x16 u; v16bf b; } c;
    c.u = *(const u16x16*)p;
    return c.b;
}

// ---------------------------------------------------------------------------
// Generic 16x16-tile WMMA GEMM: one wave per output tile, K-loop in steps of 32.
//   A: bf16 bits, row-major (lda), rows = tile rows (nodes or edges)
//   B: bf16 bits, COLUMN-major packed (ldk = K per column) -> contiguous loads
//   K remap: logical k >= ksplit maps to column k + kskip in A (concat trick)
// MODE 0: out[m*ldo+n]  = acc                                (node pre-GEMM)
// MODE 1: atomicAdd(out[dst[m]*128+n], relu(acc + bias[n] + P[src[m]*128+n]))
// MODE 2: out[m*ldo+n] += relu(acc + bias[n])                (node update)
// ---------------------------------------------------------------------------
template <int MODE>
__global__ void wmma_gemm_kernel(const unsigned short* __restrict__ A, int lda,
                                 int ksplit, int kskip,
                                 const unsigned short* __restrict__ B, int ldk,
                                 int Ktot,
                                 const float* __restrict__ bias,
                                 const float* __restrict__ P,
                                 const int* __restrict__ src,
                                 const int* __restrict__ dst,
                                 float* __restrict__ out, int ldo) {
    const int lane = threadIdx.x;               // 0..31 (wave32)
    const int m0   = blockIdx.x * 16;
    const int n0   = blockIdx.y * 16;
    const int half = lane >> 4;                 // 0 or 1
    const int rowA = m0 + (lane & 15);          // A: M = lane % 16
    const int colB = n0 + (lane & 15);          // B/D: N = lane % 16
    const int koff = half * 16;                 // A: contiguous k per half

    v8f acc = {};
    for (int kk = 0; kk < Ktot; kk += 32) {
        int k0 = kk + koff;
        int k1 = kk + 8 + koff;
        int a0 = (k0 < ksplit) ? k0 : k0 + kskip;
        int a1 = (k1 < ksplit) ? k1 : k1 + kskip;
        u16x8 ca0 = *(const u16x8*)(A + (size_t)rowA * lda + a0);
        u16x8 ca1 = *(const u16x8*)(A + (size_t)rowA * lda + a1);
        v16bf av = frag_from_chunks(ca0, ca1);
        // B 32x16: lanes 0-15 hold K=kk..kk+15, lanes 16-31 hold kk+16..kk+31
        v16bf bv = frag_from16(B + (size_t)colB * ldk + kk + half * 16);
        acc = __builtin_amdgcn_wmma_f32_16x16x32_bf16(
            false, av, false, bv, (short)0, acc, false, false);
    }

    const int mh = half * 8;                    // D: M = r + 8*(lane>=16)
#pragma unroll
    for (int r = 0; r < 8; ++r) {
        int m = m0 + r + mh;
        float v = acc[r];
        if (MODE == 0) {
            out[(size_t)m * ldo + colB] = v;
        } else if (MODE == 1) {
            int s = src[m], d = dst[m];
            v += bias[colB] + P[(size_t)s * DD + colB];
            v = v > 0.0f ? v : 0.0f;
            atomicAdd(&out[(size_t)d * DD + colB], v);
        } else {
            v += bias[colB];
            v = v > 0.0f ? v : 0.0f;
            out[(size_t)m * ldo + colB] += v;
        }
    }
}

// ---------------------------------------------------------------------------
// Elementwise / staging kernels
// ---------------------------------------------------------------------------
__global__ void zero_f32_kernel(float* p, int n) {
    int i = blockIdx.x * blockDim.x + threadIdx.x;
    if (i < n) p[i] = 0.0f;
}
__global__ void zero_i32_kernel(int* p, int n) {
    int i = blockIdx.x * blockDim.x + threadIdx.x;
    if (i < n) p[i] = 0;
}

// gather src/dst/t and full_msg -> bf16 edgefeat[:, 0:32]
__global__ void gather_edges_kernel(const int* __restrict__ ve,
                                    const int* __restrict__ full_ei,
                                    const float* __restrict__ full_t,
                                    const float* __restrict__ full_msg,
                                    int* __restrict__ src, int* __restrict__ dst,
                                    float* __restrict__ t,
                                    unsigned short* __restrict__ edgefeat) {
    int idx = blockIdx.x * blockDim.x + threadIdx.x;
    if (idx >= EV * EDIM) return;
    int e = idx >> 5, j = idx & 31;
    int eid = ve[e];
    if (j == 0) {
        src[e] = full_ei[eid];
        dst[e] = full_ei[ETOT + eid];
        t[e]   = full_t[eid];
    }
    edgefeat[(size_t)e * 160 + j] = f2bf(full_msg[(size_t)eid * EDIM + j]);
}

__global__ void hist_kernel(const int* __restrict__ src, int* __restrict__ count) {
    int i = blockIdx.x * blockDim.x + threadIdx.x;
    if (i < EV) atomicAdd(&count[src[i]], 1);
}

// single-workgroup exclusive scan over N node counts (deterministic)
__global__ void scan_kernel(const int* __restrict__ count, int* __restrict__ offset,
                            int* __restrict__ cursor, int n) {
    __shared__ int buf[1024];
    __shared__ int carry;
    if (threadIdx.x == 0) carry = 0;
    __syncthreads();
    for (int base = 0; base < n; base += 1024) {
        int i = base + (int)threadIdx.x;
        int v = (i < n) ? count[i] : 0;
        buf[threadIdx.x] = v;
        __syncthreads();
        for (int d = 1; d < 1024; d <<= 1) {
            int tv = (threadIdx.x >= (unsigned)d) ? buf[threadIdx.x - d] : 0;
            __syncthreads();
            buf[threadIdx.x] += tv;
            __syncthreads();
        }
        if (i < n) {
            offset[i] = carry + buf[threadIdx.x] - v;   // exclusive
            cursor[i] = 0;
        }
        __syncthreads();
        if (threadIdx.x == 1023) carry += buf[1023];
        __syncthreads();
    }
}

__global__ void scatter_kernel(const int* __restrict__ src,
                               const int* __restrict__ offset,
                               int* __restrict__ cursor, int* __restrict__ bucket) {
    int i = blockIdx.x * blockDim.x + threadIdx.x;
    if (i >= EV) return;
    int s = src[i];
    int p = atomicAdd(&cursor[s], 1);
    bucket[offset[s] + p] = i;
}

// per-node: sort bucket by edge index (restores stable edge order), then
// sequential inter-event delta (first event per node -> 0)
__global__ void delta_kernel(const int* __restrict__ offset,
                             const int* __restrict__ count,
                             int* __restrict__ bucket,
                             const float* __restrict__ t,
                             float* __restrict__ delta) {
    int v = blockIdx.x * blockDim.x + threadIdx.x;
    if (v >= NN) return;
    int o = offset[v], c = count[v];
    for (int i = 1; i < c; ++i) {               // insertion sort (avg bucket ~5)
        int key = bucket[o + i];
        int j = i - 1;
        while (j >= 0 && bucket[o + j] > key) { bucket[o + j + 1] = bucket[o + j]; --j; }
        bucket[o + j + 1] = key;
    }
    float prev = 0.0f;
    for (int i = 0; i < c; ++i) {
        int e = bucket[o + i];
        delta[e] = (i == 0) ? 0.0f : (t[e] - prev);
        prev = t[e];
    }
}

// time_emb -> bf16 edgefeat[:, 32:160]
__global__ void time_emb_kernel(const float* __restrict__ t,
                                const float* __restrict__ delta,
                                const float* __restrict__ wt, const float* __restrict__ bt,
                                const float* __restrict__ wd, const float* __restrict__ bd,
                                unsigned short* __restrict__ edgefeat) {
    long long idx = (long long)blockIdx.x * blockDim.x + threadIdx.x;
    if (idx >= (long long)EV * TDIM) return;
    int e = (int)(idx >> 7), j = (int)(idx & 127);
    float te = cosf(t[e] * wt[j] + bt[j]) + cosf(delta[e] * wd[j] + bd[j]);
    edgefeat[(size_t)e * 160 + 32 + j] = f2bf(te);
}

// pack all weights column-major bf16; fuse We@Wed into edge weight rows 0..31
__global__ void pack_weights_kernel(const float* __restrict__ Wx, const float* __restrict__ Wed,
                                    const float* __restrict__ Wtm, const float* __restrict__ Wmm,
                                    const float* __restrict__ Ws, const float* __restrict__ Wa,
                                    const float* __restrict__ Wv, const float* __restrict__ We,
                                    unsigned short* __restrict__ Wnode,
                                    unsigned short* __restrict__ Wupd,
                                    unsigned short* __restrict__ Uk) {
    int idx = blockIdx.x * blockDim.x + threadIdx.x;
    const int PER = 256 + 384 + 160;            // 800
    if (idx >= LAYERS * DD * PER) return;
    int l = idx / (DD * PER);
    int r = idx % (DD * PER);
    int n = r / PER;
    int kk = r % PER;
    const float* WxL  = Wx  + (size_t)l * DD * DD;
    const float* WmmL = Wmm + (size_t)l * DD * DD;
    const float* WsL  = Ws  + (size_t)l * DD * DD;
    const float* WaL  = Wa  + (size_t)l * DD * DD;
    const float* WvL  = Wv  + (size_t)l * DD * DD;
    const float* WtmL = Wtm + (size_t)l * TDIM * DD;
    const float* WedL = Wed + (size_t)l * DD * DD;
    if (kk < 256) {                             // node pre-GEMM: [Wx; Wmm]
        float v = (kk < 128) ? WxL[kk * DD + n] : WmmL[(kk - 128) * DD + n];
        Wnode[((size_t)l * DD + n) * 256 + kk] = f2bf(v);
    } else if (kk < 640) {                      // update GEMM: [Ws; Wa; Wv]
        int q = kk - 256;
        float v = (q < 128) ? WsL[q * DD + n]
                : (q < 256) ? WaL[(q - 128) * DD + n]
                            : WvL[(q - 256) * DD + n];
        Wupd[((size_t)l * DD + n) * 384 + q] = f2bf(v);
    } else {                                    // edge GEMM: [We@Wed; Wtm]
        int q = kk - 640;                       // 0..159
        float v;
        if (q < 32) {
            v = 0.0f;
            for (int d = 0; d < DD; ++d) v += We[q * DD + d] * WedL[d * DD + n];
        } else {
            v = WtmL[(q - 32) * DD + n];
        }
        Uk[((size_t)l * DD + n) * 160 + q] = f2bf(v);
    }
}

// bias_edge[l] = be @ Wed[l] + bm[l]
__global__ void fuse_bias_kernel(const float* __restrict__ be, const float* __restrict__ Wed,
                                 const float* __restrict__ bm, float* __restrict__ bias_edge) {
    int idx = blockIdx.x * blockDim.x + threadIdx.x;
    if (idx >= LAYERS * DD) return;
    int l = idx >> 7, n = idx & 127;
    const float* WedL = Wed + (size_t)l * DD * DD;
    float v = bm[l * DD + n];
    for (int d = 0; d < DD; ++d) v += be[d] * WedL[d * DD + n];
    bias_edge[idx] = v;
}

// xv = x_init; Anode[:,256:384]=bf16(evo); Anode[:,384:512]=bf16(memory)
__global__ void init_static_kernel(const float* __restrict__ x_init,
                                   const float* __restrict__ memory,
                                   const float* __restrict__ evo,
                                   float* __restrict__ xv,
                                   unsigned short* __restrict__ Anode) {
    long long idx = (long long)blockIdx.x * blockDim.x + threadIdx.x;
    if (idx >= (long long)NN * DD) return;
    int v = (int)(idx >> 7), j = (int)(idx & 127);
    xv[idx] = x_init[idx];
    Anode[(size_t)v * 512 + 256 + j] = f2bf(evo[idx]);
    Anode[(size_t)v * 512 + 384 + j] = f2bf(memory[idx]);
}

__global__ void xv_to_bf16_kernel(const float* __restrict__ xv,
                                  unsigned short* __restrict__ Anode) {
    long long idx = (long long)blockIdx.x * blockDim.x + threadIdx.x;
    if (idx >= (long long)NN * DD) return;
    int v = (int)(idx >> 7), j = (int)(idx & 127);
    Anode[(size_t)v * 512 + j] = f2bf(xv[idx]);
}

__global__ void agg_to_bf16_kernel(const float* __restrict__ agg,
                                   unsigned short* __restrict__ Anode) {
    long long idx = (long long)blockIdx.x * blockDim.x + threadIdx.x;
    if (idx >= (long long)NN * DD) return;
    int v = (int)(idx >> 7), j = (int)(idx & 127);
    Anode[(size_t)v * 512 + 128 + j] = f2bf(agg[idx]);
}

__global__ void copy_out_kernel(const float* __restrict__ xv, float* __restrict__ out, int k) {
    long long idx = (long long)blockIdx.x * blockDim.x + threadIdx.x;
    if (idx >= (long long)NN * DD) return;
    int v = (int)(idx >> 7), j = (int)(idx & 127);
    out[((size_t)v * KVIEWS + k) * DD + j] = xv[idx];
}

// ---------------------------------------------------------------------------
extern "C" void kernel_launch(void* const* d_in, const int* in_sizes, int n_in,
                              void* d_out, int out_size, void* d_ws, size_t ws_size,
                              hipStream_t stream) {
    const float* x_init   = (const float*)d_in[0];
    const int*   view_ei  = (const int*)d_in[1];
    /* d_in[2] node_view_centers: unused by reference */
    const int*   full_ei  = (const int*)d_in[3];
    const float* full_t   = (const float*)d_in[4];
    const float* full_msg = (const float*)d_in[5];
    const float* memory   = (const float*)d_in[6];
    const float* evo      = (const float*)d_in[7];
    const float* We  = (const float*)d_in[8];
    const float* be  = (const float*)d_in[9];
    const float* wt  = (const float*)d_in[10];
    const float* bt  = (const float*)d_in[11];
    const float* wd  = (const float*)d_in[12];
    const float* bd  = (const float*)d_in[13];
    const float* Wx  = (const float*)d_in[14];
    const float* Wed = (const float*)d_in[15];
    const float* Wtm = (const float*)d_in[16];
    const float* Wmm = (const float*)d_in[17];
    const float* bm  = (const float*)d_in[18];
    const float* Ws  = (const float*)d_in[19];
    const float* Wa  = (const float*)d_in[20];
    const float* Wv  = (const float*)d_in[21];
    const float* bu  = (const float*)d_in[22];
    float* out = (float*)d_out;

    // workspace bump allocator
    char* ws = (char*)d_ws;
    size_t off = 0;
    auto alloc = [&](size_t bytes) -> void* {
        void* p = ws + off;
        off = (off + bytes + 255) & ~(size_t)255;
        return p;
    };
    float*          xv       = (float*)alloc((size_t)NN * DD * 4);
    float*          P        = (float*)alloc((size_t)NN * DD * 4);
    float*          agg      = (float*)alloc((size_t)NN * DD * 4);
    unsigned short* Anode    = (unsigned short*)alloc((size_t)NN * 512 * 2);
    unsigned short* edgefeat = (unsigned short*)alloc((size_t)EV * 160 * 2);
    int*            src      = (int*)alloc((size_t)EV * 4);
    int*            dst      = (int*)alloc((size_t)EV * 4);
    float*          tbuf     = (float*)alloc((size_t)EV * 4);
    float*          delta    = (float*)alloc((size_t)EV * 4);
    int*            count    = (int*)alloc((size_t)NN * 4);
    int*            offs     = (int*)alloc((size_t)NN * 4);
    int*            cursor   = (int*)alloc((size_t)NN * 4);
    int*            bucket   = (int*)alloc((size_t)EV * 4);
    unsigned short* Wnode    = (unsigned short*)alloc((size_t)LAYERS * DD * 256 * 2);
    unsigned short* Wupd     = (unsigned short*)alloc((size_t)LAYERS * DD * 384 * 2);
    unsigned short* Uk       = (unsigned short*)alloc((size_t)LAYERS * DD * 160 * 2);
    float*          bias_e   = (float*)alloc((size_t)LAYERS * DD * 4);

    const int B256 = 256;
    const long long NDnum = (long long)NN * DD;
    const int gND  = (int)((NDnum + B256 - 1) / B256);
    const int gE32 = (EV * EDIM + B256 - 1) / B256;
    const long long E128num = (long long)EV * TDIM;
    const int gE128 = (int)((E128num + B256 - 1) / B256);
    const int gE   = (EV + B256 - 1) / B256;
    const int gN   = (NN + B256 - 1) / B256;
    const int gPW  = (LAYERS * DD * 800 + B256 - 1) / B256;
    const int BIG  = 1 << 30;

    // one-time prep (recomputed every call: deterministic, stateless)
    pack_weights_kernel<<<gPW, B256, 0, stream>>>(Wx, Wed, Wtm, Wmm, Ws, Wa, Wv, We,
                                                  Wnode, Wupd, Uk);
    fuse_bias_kernel<<<1, B256, 0, stream>>>(be, Wed, bm, bias_e);
    init_static_kernel<<<gND, B256, 0, stream>>>(x_init, memory, evo, xv, Anode);

    for (int k = 0; k < KVIEWS; ++k) {
        const int* ve = view_ei + (size_t)k * EV;
        // edge staging + deterministic inter-event deltas
        gather_edges_kernel<<<gE32, B256, 0, stream>>>(ve, full_ei, full_t, full_msg,
                                                       src, dst, tbuf, edgefeat);
        zero_i32_kernel<<<gN, B256, 0, stream>>>(count, NN);
        hist_kernel<<<gE, B256, 0, stream>>>(src, count);
        scan_kernel<<<1, 1024, 0, stream>>>(count, offs, cursor, NN);
        scatter_kernel<<<gE, B256, 0, stream>>>(src, offs, cursor, bucket);
        delta_kernel<<<gN, B256, 0, stream>>>(offs, count, bucket, tbuf, delta);
        time_emb_kernel<<<gE128, B256, 0, stream>>>(tbuf, delta, wt, bt, wd, bd, edgefeat);

        for (int l = 0; l < LAYERS; ++l) {
            // refresh xv bf16 view
            xv_to_bf16_kernel<<<gND, B256, 0, stream>>>(xv, Anode);
            // P = xv@Wx[l] + memory@Wmm[l]   (K=256 via remap 128->384)
            wmma_gemm_kernel<0><<<dim3(NN / 16, DD / 16), 32, 0, stream>>>(
                Anode, 512, 128, 256, Wnode + (size_t)l * DD * 256, 256, 256,
                nullptr, nullptr, nullptr, nullptr, P, DD);
            // agg = segment_sum(relu(P[src] + edgefeat@U[l] + bias_e[l]), dst)
            zero_f32_kernel<<<gND, B256, 0, stream>>>(agg, NN * DD);
            wmma_gemm_kernel<1><<<dim3(EV / 16, DD / 16), 32, 0, stream>>>(
                edgefeat, 160, BIG, 0, Uk + (size_t)l * DD * 160, 160, 160,
                bias_e + (size_t)l * DD, P, src, dst, agg, DD);
            // xv += relu([xv; agg; evo] @ [Ws; Wa; Wv] + bu[l])  (K=384)
            agg_to_bf16_kernel<<<gND, B256, 0, stream>>>(agg, Anode);
            wmma_gemm_kernel<2><<<dim3(NN / 16, DD / 16), 32, 0, stream>>>(
                Anode, 512, BIG, 0, Wupd + (size_t)l * DD * 384, 384, 384,
                bu + (size_t)l * DD, nullptr, nullptr, nullptr, xv, DD);
        }
        copy_out_kernel<<<gND, B256, 0, stream>>>(xv, out, k);
    }
}
